// Block_35235911696390
// MI455X (gfx1250) — matrix operand order repaired and checked by reading, not compiled, not run on hardware
//
#include <hip/hip_runtime.h>
#include <cstdint>
#include <cstddef>

typedef __bf16 bf16;
typedef __attribute__((ext_vector_type(8)))  __bf16 v8bf;
typedef __attribute__((ext_vector_type(16))) __bf16 v16bf;
typedef __attribute__((ext_vector_type(8)))  float  v8f;

#define L_SEQ   1024
#define D_MODEL 512
#define A_ATT   32
#define H_HEADS 8
#define AH      256
#define DH      4096
#define F_FFN   2048

// -------------------------------------------------------- async copy helpers
__device__ __forceinline__ void async_b128(unsigned lds_off, const void* gptr) {
  // GLOBAL_LOAD_ASYNC_TO_LDS_B128: per-lane LDS dest VGPR + 64-bit global addr.
  // Tracked with ASYNCcnt (CDNA5 ISA).
  asm volatile("global_load_async_to_lds_b128 %0, %1, off"
               :: "v"(lds_off), "v"((unsigned long long)(uintptr_t)gptr)
               : "memory");
}
__device__ __forceinline__ void wait_async0() {
  asm volatile("s_wait_asynccnt 0x0" ::: "memory");
}
__device__ __forceinline__ unsigned lds_addr(const void* p) {
  // generic LDS pointer: addr[31:0] is the wave-relative LDS byte offset
  return (unsigned)(uintptr_t)p;
}

// ------------------------------------------- f32 -> bf16 transposing convert
// src is [Kd][Nd] f32 row-major; dst is [Nd][Kd] bf16 row-major (i.e. B^T).
__global__ __launch_bounds__(256)
void cvt_transpose_bf16(const float* __restrict__ src, bf16* __restrict__ dst,
                        int Kd, int Nd) {
  __shared__ float t[32][33];
  const int kb = blockIdx.y * 32, nb = blockIdx.x * 32;
  const int tx = threadIdx.x & 31, ty = threadIdx.x >> 5;  // ty: 0..7
#pragma unroll
  for (int i = ty; i < 32; i += 8)
    t[i][tx] = src[(size_t)(kb + i) * Nd + nb + tx];
  __syncthreads();
#pragma unroll
  for (int i = ty; i < 32; i += 8)
    dst[(size_t)(nb + i) * Kd + kb + tx] = (bf16)t[tx][i];
}

// ---------------------------------------------------------------- LayerNorm 0
__global__ __launch_bounds__(256)
void ln0_kernel(const float* __restrict__ x, const float* __restrict__ w,
                const float* __restrict__ b, bf16* __restrict__ lnxb,
                bf16* __restrict__ xb) {
  const int l = blockIdx.x, t = threadIdx.x;
  __shared__ float red[256];
  __shared__ float stat[2];
  const float v0 = x[l * D_MODEL + t];
  const float v1 = x[l * D_MODEL + 256 + t];
  red[t] = v0 + v1;
  __syncthreads();
  for (int s = 128; s > 0; s >>= 1) { if (t < s) red[t] += red[t + s]; __syncthreads(); }
  if (t == 0) stat[0] = red[0] * (1.0f / D_MODEL);
  __syncthreads();
  const float mu = stat[0];
  const float d0 = v0 - mu, d1 = v1 - mu;
  red[t] = d0 * d0 + d1 * d1;
  __syncthreads();
  for (int s = 128; s > 0; s >>= 1) { if (t < s) red[t] += red[t + s]; __syncthreads(); }
  if (t == 0) stat[1] = rsqrtf(red[0] * (1.0f / D_MODEL) + 1e-5f);
  __syncthreads();
  const float rstd = stat[1];
  lnxb[l * D_MODEL + t]       = (bf16)(d0 * rstd * w[t] + b[t]);
  lnxb[l * D_MODEL + 256 + t] = (bf16)(d1 * rstd * w[256 + t] + b[256 + t]);
  xb[l * D_MODEL + t]       = (bf16)v0;
  xb[l * D_MODEL + 256 + t] = (bf16)v1;
}

// ---------------------------------------------------------------- WMMA GEMM
// C[M,N] = act(A[M,K] @ B[K,N] + bias) (+resid).
// A bf16 [M][K]; BT bf16 [N][K] (pre-transposed weights).
// Block tile 128x128, K-step 64, 256 threads = 8 waves, wave tile 32x64.
// LDS double-buffered; tiles staged with global_load_async_to_lds_b128.
__device__ __forceinline__ v16bf cat16(v8bf lo, v8bf hi) {
  return __builtin_shufflevector(lo, hi, 0, 1, 2, 3, 4, 5, 6, 7,
                                 8, 9, 10, 11, 12, 13, 14, 15);
}

__global__ __launch_bounds__(256)
void gemm_bf16_kernel(const bf16* __restrict__ A, const bf16* __restrict__ BT,
                      const float* __restrict__ bias, float* __restrict__ outF,
                      bf16* __restrict__ outB, const float* __restrict__ resid,
                      int M, int N, int K, int act) {
  __shared__ __attribute__((aligned(16))) bf16 sA[2][128][72];   // [buf][m][k]
  __shared__ __attribute__((aligned(16))) bf16 sBT[2][128][72];  // [buf][n][k]
  const int tid  = threadIdx.x;
  const int bm   = blockIdx.y * 128;
  const int bn   = blockIdx.x * 128;
  const int wave = tid >> 5;
  const int lane = tid & 31;
  const int l16  = lane & 15;
  const int hf   = (lane >> 4) & 1;  // lane half (ISA 16-bit frag layouts)
  const int wy   = wave >> 1;        // 0..3: 32-row slab
  const int wx   = wave & 1;         // 0..1: 64-col slab

  v8f acc[2][4];
#pragma unroll
  for (int mi = 0; mi < 2; ++mi)
#pragma unroll
    for (int ni = 0; ni < 4; ++ni)
      acc[mi][ni] = (v8f){0.f, 0.f, 0.f, 0.f, 0.f, 0.f, 0.f, 0.f};

  const int stRow = tid >> 1;        // 0..127: tile row this thread stages
  const int stK   = (tid & 1) << 5;  // 0 | 32 halfs within the 64-wide k slab

  // issue one 128x64 A tile + 128x64 B^T tile into LDS buffer `buf`
  auto issue_tile = [&](int buf, int k0) {
    const bf16* gA = A + (size_t)(bm + stRow) * K + k0 + stK;
    unsigned la = lds_addr(&sA[buf][stRow][stK]);
    async_b128(la,      gA);
    async_b128(la + 16, gA + 8);
    async_b128(la + 32, gA + 16);
    async_b128(la + 48, gA + 24);
    const bf16* gB = BT + (size_t)(bn + stRow) * K + k0 + stK;
    unsigned lb = lds_addr(&sBT[buf][stRow][stK]);
    async_b128(lb,      gB);
    async_b128(lb + 16, gB + 8);
    async_b128(lb + 32, gB + 16);
    async_b128(lb + 48, gB + 24);
  };

  issue_tile(0, 0);
  int cur = 0;
  for (int k0 = 0; k0 < K; k0 += 64) {
    wait_async0();        // own async writes into `cur` complete
    __syncthreads();      // all waves: writes visible, prior reads drained
    if (k0 + 64 < K) issue_tile(cur ^ 1, k0 + 64);

#pragma unroll
    for (int ks = 0; ks < 2; ++ks) {
      v16bf af[2], bfr[4];
#pragma unroll
      for (int mi = 0; mi < 2; ++mi) {
        // A frag: elems 0-7 <- K = hf*8+0..7 ; elems 8-15 <- K = 16+hf*8+0..7
        const bf16* p = &sA[cur][wy * 32 + mi * 16 + l16][ks * 32];
        v8bf lo = *(const v8bf*)(p + hf * 8);
        v8bf hi = *(const v8bf*)(p + 16 + hf * 8);
        af[mi] = cat16(lo, hi);
      }
#pragma unroll
      for (int ni = 0; ni < 4; ++ni) {
        // B frag: elems 0-7 <- K = hf*8+0..7 ; elems 8-15 <- K = 16+hf*8+0..7, col N = lane%16
        const bf16* p = &sBT[cur][wx * 64 + ni * 16 + l16][ks * 32];
        v8bf lo = *(const v8bf*)(p + hf * 8);
        v8bf hi = *(const v8bf*)(p + 16 + hf * 8);
        bfr[ni] = cat16(lo, hi);
      }
#pragma unroll
      for (int mi = 0; mi < 2; ++mi)
#pragma unroll
        for (int ni = 0; ni < 4; ++ni)
          acc[mi][ni] = __builtin_amdgcn_wmma_f32_16x16x32_bf16(
              false, af[mi], false, bfr[ni], (short)0, acc[mi][ni], false, false);
    }
    cur ^= 1;
  }

  // epilogue: C row = hf*8 + r, col = lane%16 within each 16x16 tile
#pragma unroll
  for (int mi = 0; mi < 2; ++mi) {
#pragma unroll
    for (int ni = 0; ni < 4; ++ni) {
      const int col  = bn + wx * 64 + ni * 16 + l16;
      const int row0 = bm + wy * 32 + mi * 16 + hf * 8;
      const float bc = bias ? bias[col] : 0.0f;
#pragma unroll
      for (int r = 0; r < 8; ++r) {
        float val = acc[mi][ni][r] + bc;
        if (act == 1)      val = fmaxf(val, 0.0f);
        else if (act == 2) val = expf(fminf(fmaxf(val, -10.0f), 10.0f));
        const size_t idx = (size_t)(row0 + r) * N + col;
        if (resid) val += resid[idx];
        if (outF) outF[idx] = val;
        if (outB) outB[idx] = (bf16)val;
      }
    }
  }
}

// -------------------------------------------- linear-attention register scan
__global__ __launch_bounds__(64)
void attn_scan_kernel(const float* __restrict__ q, const float* __restrict__ k,
                      const float* __restrict__ v, float* __restrict__ qkv) {
  const int h = blockIdx.x >> 3;
  const int d = (blockIdx.x & 7) * 64 + threadIdx.x;
  __shared__ float sK[16][32];
  __shared__ float sQ[16][32];
  float S[A_ATT];
#pragma unroll
  for (int a = 0; a < A_ATT; ++a) S[a] = 0.0f;

  for (int lc = 0; lc < L_SEQ; lc += 16) {
    __syncthreads();
#pragma unroll
    for (int i = 0; i < 16; ++i) {
      const int t = threadIdx.x;
      const int l = lc + i;
      if (t < 32) sK[i][t]      = k[l * AH + t * H_HEADS + h];
      else        sQ[i][t - 32] = q[l * AH + (t - 32) * H_HEADS + h];
    }
    __syncthreads();
#pragma unroll 4
    for (int i = 0; i < 16; ++i) {
      const int l = lc + i;
      const float vv = v[(size_t)l * DH + d * H_HEADS + h];
      float o = 0.0f;
#pragma unroll
      for (int a = 0; a < A_ATT; ++a) {
        S[a] = __builtin_fmaf(sK[i][a], vv, S[a]);   // S += k_l v_l^T (inclusive)
        o    = __builtin_fmaf(sQ[i][a], S[a], o);    // qkv_l = q_l . S_l
      }
      qkv[(size_t)l * DH + d * H_HEADS + h] = o;
    }
  }
}

// -------------------------------------------- den[l,h] = q_l . cumsum(k)_l
__global__ __launch_bounds__(32)
void den_kernel(const float* __restrict__ q, const float* __restrict__ k,
                float* __restrict__ den) {
  const int h = blockIdx.x;
  const int a = threadIdx.x;
  float kc = 0.0f;
  for (int l = 0; l < L_SEQ; ++l) {
    kc += k[l * AH + a * H_HEADS + h];
    float p = q[l * AH + a * H_HEADS + h] * kc;
#pragma unroll
    for (int off = 16; off > 0; off >>= 1) p += __shfl_xor(p, off, 32);
    if (a == 0) den[l * H_HEADS + h] = p;
  }
}

// ---------------------------------------------------------------- LayerNorm 1
__global__ __launch_bounds__(256)
void ln1_kernel(const float* __restrict__ qkv, const float* __restrict__ den,
                const float* __restrict__ x, const float* __restrict__ w,
                const float* __restrict__ b, bf16* __restrict__ yb) {
  const int l = blockIdx.x, t = threadIdx.x;
  __shared__ float red[256];
  __shared__ float stat[2];
  float z[16];
  float s = 0.0f;
#pragma unroll
  for (int i = 0; i < 16; ++i) {
    const int e = t * 16 + i;
    z[i] = qkv[(size_t)l * DH + e] / den[l * H_HEADS + (e & 7)];
    s += z[i];
  }
  red[t] = s;
  __syncthreads();
  for (int st = 128; st > 0; st >>= 1) { if (t < st) red[t] += red[t + st]; __syncthreads(); }
  if (t == 0) stat[0] = red[0] * (1.0f / DH);
  __syncthreads();
  const float mu = stat[0];
  s = 0.0f;
#pragma unroll
  for (int i = 0; i < 16; ++i) { const float dd = z[i] - mu; s += dd * dd; }
  red[t] = s;
  __syncthreads();
  for (int st = 128; st > 0; st >>= 1) { if (t < st) red[t] += red[t + st]; __syncthreads(); }
  if (t == 0) stat[1] = rsqrtf(red[0] * (1.0f / DH) + 1e-5f);
  __syncthreads();
  const float rstd = stat[1];
#pragma unroll
  for (int i = 0; i < 16; ++i) {
    const int e = t * 16 + i;
    const int d = e >> 3;
    const float y = x[l * D_MODEL + d] + (z[i] - mu) * rstd * w[e] + b[e];
    yb[(size_t)l * DH + e] = (bf16)y;
  }
}

// ================================================================= launcher
extern "C" void kernel_launch(void* const* d_in, const int* in_sizes, int n_in,
                              void* d_out, int out_size, void* d_ws, size_t ws_size,
                              hipStream_t stream) {
  (void)in_sizes; (void)n_in; (void)out_size; (void)ws_size;
  const float* x    = (const float*)d_in[0];
  const float* ln0w = (const float*)d_in[1];
  const float* ln0b = (const float*)d_in[2];
  const float* ln1w = (const float*)d_in[3];
  const float* ln1b = (const float*)d_in[4];
  const float* qw   = (const float*)d_in[5];
  const float* qb   = (const float*)d_in[6];
  const float* kw   = (const float*)d_in[7];
  const float* kb   = (const float*)d_in[8];
  const float* vw   = (const float*)d_in[9];
  const float* vb   = (const float*)d_in[10];
  const float* w1   = (const float*)d_in[11];
  const float* b1   = (const float*)d_in[12];
  const float* w2   = (const float*)d_in[13];
  const float* b2   = (const float*)d_in[14];
  float* out = (float*)d_out;

  char* ws = (char*)d_ws;
  size_t off = 0;
  auto take = [&](size_t bytes) -> char* {
    char* p = ws + off;
    off = (off + bytes + 255) & ~(size_t)255;
    return p;
  };
  bf16*  lnxb = (bf16*)take((size_t)L_SEQ * D_MODEL * 2);
  bf16*  xb   = (bf16*)take((size_t)L_SEQ * D_MODEL * 2);
  bf16*  qwT  = (bf16*)take((size_t)AH * D_MODEL * 2);      // [N][K]
  bf16*  kwT  = (bf16*)take((size_t)AH * D_MODEL * 2);
  bf16*  vwT  = (bf16*)take((size_t)DH * D_MODEL * 2);
  bf16*  w1T  = (bf16*)take((size_t)F_FFN * DH * 2);
  bf16*  w2T  = (bf16*)take((size_t)D_MODEL * F_FFN * 2);
  float* qf   = (float*)take((size_t)L_SEQ * AH * 4);
  float* kf   = (float*)take((size_t)L_SEQ * AH * 4);
  float* vf   = (float*)take((size_t)L_SEQ * DH * 4);
  float* qkvf = (float*)take((size_t)L_SEQ * DH * 4);
  float* denf = (float*)take((size_t)L_SEQ * H_HEADS * 4);
  bf16*  yb   = (bf16*)take((size_t)L_SEQ * DH * 2);
  bf16*  h1b  = (bf16*)take((size_t)L_SEQ * F_FFN * 2);

  auto cvtT = [&](const float* s, bf16* dst, int K, int N) {
    cvt_transpose_bf16<<<dim3(N / 32, K / 32), 256, 0, stream>>>(s, dst, K, N);
  };
  cvtT(qw, qwT, D_MODEL, AH);
  cvtT(kw, kwT, D_MODEL, AH);
  cvtT(vw, vwT, D_MODEL, DH);
  cvtT(w1, w1T, DH, F_FFN);
  cvtT(w2, w2T, F_FFN, D_MODEL);

  ln0_kernel<<<L_SEQ, 256, 0, stream>>>(x, ln0w, ln0b, lnxb, xb);

  // q = exp(clip(_x @ qw + qb)), k likewise, v = x @ vw + vb
  gemm_bf16_kernel<<<dim3(AH / 128, L_SEQ / 128), 256, 0, stream>>>(
      lnxb, qwT, qb, qf, nullptr, nullptr, L_SEQ, AH, D_MODEL, /*act=*/2);
  gemm_bf16_kernel<<<dim3(AH / 128, L_SEQ / 128), 256, 0, stream>>>(
      lnxb, kwT, kb, kf, nullptr, nullptr, L_SEQ, AH, D_MODEL, /*act=*/2);
  gemm_bf16_kernel<<<dim3(DH / 128, L_SEQ / 128), 256, 0, stream>>>(
      xb, vwT, vb, vf, nullptr, nullptr, L_SEQ, DH, D_MODEL, /*act=*/0);

  den_kernel<<<H_HEADS, 32, 0, stream>>>(qf, kf, denf);
  attn_scan_kernel<<<H_HEADS * (D_MODEL / 64), 64, 0, stream>>>(qf, kf, vf, qkvf);

  ln1_kernel<<<L_SEQ, 256, 0, stream>>>(qkvf, denf, x, ln1w, ln1b, yb);

  // h1 = relu(y @ w1 + b1) -> bf16 ; out = relu(h1 @ w2 + b2) + x -> f32
  gemm_bf16_kernel<<<dim3(F_FFN / 128, L_SEQ / 128), 256, 0, stream>>>(
      yb, w1T, b1, nullptr, h1b, nullptr, L_SEQ, F_FFN, DH, /*act=*/1);
  gemm_bf16_kernel<<<dim3(D_MODEL / 128, L_SEQ / 128), 256, 0, stream>>>(
      h1b, w2T, b2, out, nullptr, x, L_SEQ, D_MODEL, F_FFN, /*act=*/1);
}
